// Attention_19387482375108
// MI455X (gfx1250) — matrix operand order repaired and checked by reading, not compiled, not run on hardware
//
#include <hip/hip_runtime.h>

#define B_ 8
#define T_ 2048
#define D_ 1024
#define H_ 128

typedef __attribute__((ext_vector_type(16))) __bf16 v16bf;
typedef __attribute__((ext_vector_type(8)))  __bf16 v8bf;
typedef __attribute__((ext_vector_type(8)))  float  v8f;
typedef unsigned int u32x4 __attribute__((ext_vector_type(4)));
typedef int          i32x4 __attribute__((ext_vector_type(4)));
typedef int          i32x8 __attribute__((ext_vector_type(8)));

union V16 { v16bf v; v8bf h[2]; };

#if defined(__has_builtin)
#  if __has_builtin(__builtin_amdgcn_tensor_load_to_lds)
#    define HAVE_TDM 1
#  endif
#endif
#ifndef HAVE_TDM
#  define HAVE_TDM 0
#endif

__device__ __forceinline__ v8f wmma_bf16(v16bf a, v16bf b, v8f c) {
    return __builtin_amdgcn_wmma_f32_16x16x32_bf16(
        false, a, false, b, (short)0, c, false, false);
}

#if HAVE_TDM
// Issue one TDM 2-D tile load (bf16 elements) global -> LDS.
// D# group0: count=1 | lds_addr | global_addr | type=2
// D# group1: data_size=2B | tensor_dim0/1 == tile_dim0/1 (tile fully in-bounds) | dim0 stride
__device__ __forceinline__ void tdm_load_tile_2d(unsigned lds_off, const void* gptr,
                                                 unsigned tile_d0, unsigned tile_d1,
                                                 unsigned stride0_elems)
{
    unsigned long long ga = (unsigned long long)(uintptr_t)gptr;
    u32x4 g0;
    g0[0] = 1u;                                                  // count=1, user descriptor
    g0[1] = lds_off;                                             // lds_addr (bytes)
    g0[2] = (unsigned)(ga & 0xffffffffu);                        // global_addr[31:0]
    g0[3] = (unsigned)((ga >> 32) & 0x01ffffffu) | (2u << 30);   // global_addr[56:32], type=2
    i32x8 g1;
    g1[0] = (int)(1u << 16);            // workgroup_mask=0, data_size=1 (2 bytes)
    g1[1] = (int)(tile_d0 << 16);       // tensor_dim0[15:0] -> bits[63:48]
    g1[2] = (int)(tile_d1 << 16);       // tensor_dim1[15:0] -> bits[95:80]
    g1[3] = (int)(tile_d0 << 16);       // tile_dim0        -> bits[127:112]
    g1[4] = (int)tile_d1;               // tile_dim1        -> bits[143:128], tile_dim2=0
    g1[5] = (int)stride0_elems;         // tensor_dim0_stride[31:0]
    g1[6] = 0;
    g1[7] = 0;
    i32x4 z4 = {0, 0, 0, 0};
#if __clang_major__ >= 23
    i32x8 z8 = {0, 0, 0, 0, 0, 0, 0, 0};
    __builtin_amdgcn_tensor_load_to_lds(g0, g1, z4, z4, z8, 0);
#else
    __builtin_amdgcn_tensor_load_to_lds(g0, g1, z4, z4, 0);
#endif
}
#endif

// ---------------------------------------------------------------------------
// Pass 1a: x fp32 -> bf16, elementwise (bandwidth-bound, ~96MB total traffic).
// ---------------------------------------------------------------------------
__global__ __launch_bounds__(256) void cvt_x_kernel(const float* __restrict__ x,
                                                    __bf16* __restrict__ xb)
{
    const size_t i = ((size_t)blockIdx.x * 256 + threadIdx.x) * 8;
    const float4 a = *(const float4*)(x + i);
    const float4 c = *(const float4*)(x + i + 4);
    v8bf o;
    o[0] = (__bf16)a.x; o[1] = (__bf16)a.y; o[2] = (__bf16)a.z; o[3] = (__bf16)a.w;
    o[4] = (__bf16)c.x; o[5] = (__bf16)c.y; o[6] = (__bf16)c.z; o[7] = (__bf16)c.w;
    *(v8bf*)(xb + i) = o;
}

// ---------------------------------------------------------------------------
// Pass 1b: Wq/Wk/Wv fp32 [D][H] -> bf16 transposed Wt[3][H][D] so the GEMM
// B-fragment is one contiguous 32B load. 1.5MB source, L2-resident.
// ---------------------------------------------------------------------------
__global__ __launch_bounds__(256) void cvt_w_kernel(const float* __restrict__ Wq,
                                                    const float* __restrict__ Wk,
                                                    const float* __restrict__ Wv,
                                                    __bf16* __restrict__ Wt)
{
    const int tid = blockIdx.x * 256 + threadIdx.x;   // 3 * 128 * 128 = 49152
    const int dc  = tid & 127;          // D/8 chunks
    const int h   = (tid >> 7) & 127;   // H
    const int m   = tid >> 14;          // matrix 0..2
    const float* W = (m == 0) ? Wq : (m == 1) ? Wk : Wv;
    v8bf o;
    #pragma unroll
    for (int e = 0; e < 8; ++e) o[e] = (__bf16)W[(size_t)(dc * 8 + e) * H_ + h];
    *(v8bf*)(Wt + ((size_t)m * H_ + h) * D_ + dc * 8) = o;
}

// ---------------------------------------------------------------------------
// Pass 2: QKV projection, all-bf16 operands, fp32 WMMA accumulation.
// One wave -> one 16x16 tile. V stored transposed VT[B][H][T].
// ---------------------------------------------------------------------------
__global__ __launch_bounds__(128) void qkv_proj_kernel(
    const __bf16* __restrict__ xb, const __bf16* __restrict__ Wt,
    __bf16* __restrict__ Qb, __bf16* __restrict__ Kb, __bf16* __restrict__ VTb)
{
    const int lane   = threadIdx.x & 31;
    const int lane16 = lane & 15;
    const bool hi    = lane >= 16;

    const int wid   = blockIdx.x * 4 + (threadIdx.x >> 5);
    const int hTile = wid & 7;
    const int tTile = (wid >> 3) & 127;
    const int b     = (wid >> 10) & 7;
    const int mat   = wid >> 13;

    const int t0 = tTile * 16, h0 = hTile * 16;
    const __bf16* xrow = xb + (size_t)(b * T_ + t0 + lane16) * D_ + (hi ? 16 : 0);
    const __bf16* wrow = Wt + ((size_t)mat * H_ + h0 + lane16) * D_ + (hi ? 16 : 0);

    v8f acc = {0.f,0.f,0.f,0.f,0.f,0.f,0.f,0.f};
    for (int ks = 0; ks < D_ / 32; ++ks) {
        V16 a, bm;
        a.h[0] = *(const v8bf*)(xrow + ks * 32);
        a.h[1] = *(const v8bf*)(xrow + ks * 32 + 8);
        bm.v   = *(const v16bf*)(wrow + ks * 32);
        acc = wmma_bf16(a.v, bm.v, acc);
    }

    const int rowbase = hi ? 8 : 0;
    if (mat == 2) {
        #pragma unroll
        for (int r = 0; r < 8; ++r)
            VTb[((size_t)b * H_ + h0 + lane16) * T_ + t0 + rowbase + r] = (__bf16)acc[r];
    } else {
        __bf16* dst = (mat == 0) ? Qb : Kb;
        #pragma unroll
        for (int r = 0; r < 8; ++r)
            dst[(size_t)(b * T_ + t0 + rowbase + r) * H_ + h0 + lane16] = (__bf16)acc[r];
    }
}

// ---------------------------------------------------------------------------
// Pass 3: causal flash attention. One wave per workgroup (descriptors stay
// wave-uniform). K/V^T tiles staged to LDS by the Tensor Data Mover with
// double buffering; compute on block j overlaps TDM of block j+1.
// ---------------------------------------------------------------------------
__global__ __launch_bounds__(32) void flash_attn_kernel(
    const __bf16* __restrict__ Qb, const __bf16* __restrict__ Kb,
    const __bf16* __restrict__ VTb, float* __restrict__ out)
{
    __shared__ __bf16 sK[2][32][128];   // key block, row-major  (2 x 8KB)
    __shared__ __bf16 sV[2][128][32];   // V^T block, h-major    (2 x 8KB)
    __shared__ __bf16 sP[16][32];       // probability tile      (1KB)

    const int lane    = threadIdx.x;
    const int lane16  = lane & 15;
    const bool hi     = lane >= 16;
    const int rowbase = hi ? 8 : 0;

    const int qTile = blockIdx.x & 127;
    const int b     = blockIdx.x >> 7;
    const int q0    = qTile * 16;

    // Stage one 32-key block (buf selects double buffer).
    auto stage = [&](int buf, int k0) {
#if HAVE_TDM
        tdm_load_tile_2d((unsigned)(uintptr_t)&sK[buf][0][0],
                         Kb + (size_t)(b * T_ + k0) * H_, H_, 32, H_);
        tdm_load_tile_2d((unsigned)(uintptr_t)&sV[buf][0][0],
                         VTb + (size_t)b * H_ * T_ + k0, 32, H_, T_);
#else
        #pragma unroll
        for (int c = 0; c < 8; ++c)
            *(v16bf*)&sK[buf][lane][c * 16] =
                *(const v16bf*)(Kb + (size_t)(b * T_ + k0 + lane) * H_ + c * 16);
        #pragma unroll
        for (int r = 0; r < 4; ++r)
            #pragma unroll
            for (int c = 0; c < 2; ++c)
                *(v16bf*)&sV[buf][lane * 4 + r][c * 16] =
                    *(const v16bf*)(VTb + (size_t)(b * H_ + lane * 4 + r) * T_ + k0 + c * 16);
#endif
    };

    // Q fragments (A-matrix layout), resident in VGPRs.
    V16 qf[4];
    const __bf16* qrow = Qb + (size_t)(b * T_ + q0 + lane16) * H_;
    #pragma unroll
    for (int f = 0; f < 4; ++f) {
        const int kb = f * 32 + (hi ? 16 : 0);
        qf[f].h[0] = *(const v8bf*)(qrow + kb);
        qf[f].h[1] = *(const v8bf*)(qrow + kb + 8);
    }

    float m_i[8], l_i[8];
    v8f O[8];
    const v8f zero8 = {0.f,0.f,0.f,0.f,0.f,0.f,0.f,0.f};
    #pragma unroll
    for (int r = 0; r < 8; ++r) { m_i[r] = -1e30f; l_i[r] = 0.f; }
    #pragma unroll
    for (int t = 0; t < 8; ++t) O[t] = zero8;

    const float scale = 0.088388347648318447f;   // 1/sqrt(H)

    stage(0, 0);
    int it = 0;
    for (int k0 = 0; k0 <= q0 + 15; k0 += 32, ++it) {
        const int  buf     = it & 1;
        const bool hasNext = (k0 + 32) <= (q0 + 15);
#if HAVE_TDM
        if (hasNext) {
            stage(buf ^ 1, k0 + 32);                 // prefetch next block
            __builtin_amdgcn_s_wait_tensorcnt(2);    // current block's 2 ops done
        } else {
            __builtin_amdgcn_s_wait_tensorcnt(0);
        }
        asm volatile("" ::: "memory");
#else
        if (hasNext) stage(buf ^ 1, k0 + 32);
        asm volatile("s_wait_dscnt 0" ::: "memory");
#endif

        // ---- S = Q * K^T from LDS key tile ----
        v8f S[2] = { zero8, zero8 };
        #pragma unroll
        for (int s = 0; s < 2; ++s) {
            v8f a = S[s];
            #pragma unroll
            for (int f = 0; f < 4; ++f) {
                V16 bk;
                bk.v = *(const v16bf*)&sK[buf][s * 16 + lane16][f * 32 + (hi ? 16 : 0)];
                a = wmma_bf16(qf[f].v, bk.v, a);
            }
            S[s] = a;
        }

        // ---- scale, causal mask, online softmax ----
        const bool edge = (k0 + 31 > q0);
        #pragma unroll
        for (int r = 0; r < 8; ++r) {
            const int q = q0 + rowbase + r;
            float a0 = S[0][r] * scale;
            float a1 = S[1][r] * scale;
            if (edge) {
                if (k0 + lane16 > q)      a0 = -1e30f;
                if (k0 + 16 + lane16 > q) a1 = -1e30f;
            }
            float mx = fmaxf(a0, a1);
            #pragma unroll
            for (int off = 1; off < 16; off <<= 1)
                mx = fmaxf(mx, __shfl_xor(mx, off, 16));
            const float mn = fmaxf(m_i[r], mx);
            const float al = __expf(m_i[r] - mn);
            const float p0 = __expf(a0 - mn);
            const float p1 = __expf(a1 - mn);
            float rs = p0 + p1;
            #pragma unroll
            for (int off = 1; off < 16; off <<= 1)
                rs += __shfl_xor(rs, off, 16);
            l_i[r] = l_i[r] * al + rs;
            m_i[r] = mn;
            sP[rowbase + r][lane16]      = (__bf16)p0;
            sP[rowbase + r][lane16 + 16] = (__bf16)p1;
            #pragma unroll
            for (int t = 0; t < 8; ++t) O[t][r] *= al;
        }

        asm volatile("s_wait_dscnt 0" ::: "memory");   // P stores visible

        V16 pa;
        {
            const int ah = hi ? 16 : 0;
            pa.h[0] = *(const v8bf*)&sP[lane16][ah];
            pa.h[1] = *(const v8bf*)&sP[lane16][ah + 8];
        }

        // ---- O += P * V from LDS V^T tile, exact K=32 ----
        #pragma unroll
        for (int t = 0; t < 8; ++t) {
            V16 bv;
            bv.v = *(const v16bf*)&sV[buf][t * 16 + lane16][hi ? 16 : 0];
            O[t] = wmma_bf16(pa.v, bv.v, O[t]);
        }
        asm volatile("s_wait_dscnt 0" ::: "memory");   // WAR guard on sP
    }

    // ---- epilogue: normalize and store fp32 ----
    #pragma unroll
    for (int t = 0; t < 8; ++t) {
        #pragma unroll
        for (int r = 0; r < 8; ++r) {
            out[(size_t)(b * T_ + q0 + rowbase + r) * H_ + t * 16 + lane16] =
                O[t][r] / l_i[r];
        }
    }
}

extern "C" void kernel_launch(void* const* d_in, const int* in_sizes, int n_in,
                              void* d_out, int out_size, void* d_ws, size_t ws_size,
                              hipStream_t stream)
{
    const float* x  = (const float*)d_in[0];
    const float* Wq = (const float*)d_in[1];
    const float* Wk = (const float*)d_in[2];
    const float* Wv = (const float*)d_in[3];
    float* out = (float*)d_out;

    // Workspace layout (bf16 elements), all 32B-aligned:
    __bf16* xb  = (__bf16*)d_ws;                       // B*T*D   = 16,777,216
    __bf16* Wtb = xb  + (size_t)B_ * T_ * D_;          // 3*H*D   =    393,216
    __bf16* Qb  = Wtb + (size_t)3 * H_ * D_;           // B*T*H   =  2,097,152
    __bf16* Kb  = Qb  + (size_t)B_ * T_ * H_;
    __bf16* VTb = Kb  + (size_t)B_ * T_ * H_;

    cvt_x_kernel<<<dim3(8192), dim3(256), 0, stream>>>(x, xb);
    cvt_w_kernel<<<dim3(192),  dim3(256), 0, stream>>>(Wq, Wk, Wv, Wtb);
    qkv_proj_kernel<<<dim3(6144), dim3(128), 0, stream>>>(xb, Wtb, Qb, Kb, VTb);
    flash_attn_kernel<<<dim3(1024), dim3(32), 0, stream>>>(Qb, Kb, VTb, out);
}
